// QuantumNeuralNetwork_13013750907311
// MI455X (gfx1250) — matrix operand order repaired and checked by reading, not compiled, not run on hardware
//
#include <hip/hip_runtime.h>
#include <hip/hip_bf16.h>

typedef __attribute__((ext_vector_type(16))) _Float16 v16h;
typedef __attribute__((ext_vector_type(8)))  _Float16 v8h;
typedef __attribute__((ext_vector_type(8)))  float    v8f;
typedef __attribute__((ext_vector_type(2)))  float    v2f;
typedef __attribute__((ext_vector_type(4)))  unsigned int u32x4;
typedef __attribute__((ext_vector_type(8)))  int      i32x8;
typedef __attribute__((ext_vector_type(4)))  int      i32x4;

union F16Frag { v16h v; v8h h[2]; };

#define DIM 512
#define BM 64
#define BN 64
#define BK 64
#define LDA 72   // halves; 144B row stride: 16B aligned, conflict-free mod 64 banks
#define LDB 72

// ---------------- TDM support (toolchain-portable guards) -------------------
#if __has_builtin(__builtin_amdgcn_tensor_load_to_lds)
#define HAVE_TDM 1
#if __has_include(<hip/amd_detail/amd_gfx1250_TDM.h>)
#warning "QNN marker: TDM path = 6-arg (amdgpu-toolchain headers present)"
#define TDM_LOAD(g0, g1, z4a, z4b, z8, cpol) \
  __builtin_amdgcn_tensor_load_to_lds((g0), (g1), (z4a), (z4b), (z8), (cpol))
#else
#warning "QNN marker: TDM path = 5-arg (ROCm 7.2 style)"
#define TDM_LOAD(g0, g1, z4a, z4b, z8, cpol) \
  __builtin_amdgcn_tensor_load_to_lds((g0), (g1), (z4a), (z4b), (cpol))
#endif
#else
#warning "QNN marker: TDM builtin MISSING - synchronous fallback copies in use"
#endif

#if __has_builtin(__builtin_amdgcn_s_wait_tensorcnt)
#define WAIT_TENSORCNT0() __builtin_amdgcn_s_wait_tensorcnt(0)
#else
#warning "QNN marker: s_wait_tensorcnt builtin MISSING - inline asm fallback"
#define WAIT_TENSORCNT0() asm volatile("s_wait_tensorcnt 0x0" ::: "memory")
#endif

// Low 32 bits of a generic pointer to LDS == byte offset within the
// workgroup's LDS allocation (aperture layout, ISA ch.10.2).
__device__ __forceinline__ unsigned int lds_offset(const void* p) {
  return (unsigned int)(uintptr_t)p;
}

#if defined(HAVE_TDM)
// Issue one 2-D TDM tile load: `rows` rows x 64 halves from a 512-wide f16
// matrix into LDS with 8-half row padding (72-half effective stride).
__device__ __forceinline__ void tdm_tile_load(unsigned int lds_off,
                                              const _Float16* gsrc,
                                              int rows) {
  unsigned long long ga = (unsigned long long)(uintptr_t)gsrc;
  u32x4 g0 = {1u,                       // count=1, user descriptor
              lds_off,                  // lds_addr
              (unsigned int)ga,         // global_addr[31:0]
              (unsigned int)((ga >> 32) & 0x01FFFFFFu) | (2u << 30)}; // type=2
  i32x8 g1 = {(int)((1u << 16)      // data_size = 2 bytes
                    | (1u << 20)    // pad_enable
                    | (4u << 22)    // pad_interval: 32 dwords (=128B row)
                    | (3u << 25)),  // pad_amount:   4 dwords (=16B = 8 halves)
              (int)(512u << 16),    // tensor_dim0[15:0] @ bits[63:48]
              (int)(512u << 16),    // tensor_dim1[15:0] @ bits[95:80]
              (int)(64u << 16),     // tile_dim0 = 64 halves @ bits[127:112]
              rows,                 // tile_dim1 = rows @ bits[143:128]
              512,                  // tensor_dim0_stride lo
              0, 0};
  i32x4 z4 = {0, 0, 0, 0};
  i32x8 z8 = {0, 0, 0, 0, 0, 0, 0, 0};
  TDM_LOAD(g0, g1, z4, z4, z8, 0);
}
#endif

// ---------------------------------------------------------------------------
// Kernel 1: build U (transposed) by simulating the circuit on basis vectors.
// Urt[j*512 + i] = U[i][j] (column j of U, stored contiguously).
// ---------------------------------------------------------------------------
__global__ __launch_bounds__(512) void qnn_buildU(const float* __restrict__ W,
                                                  float* __restrict__ Urt_r,
                                                  float* __restrict__ Urt_i) {
  __shared__ float sr[DIM], si[DIM];
  __shared__ float G[27][8];
  const int i = threadIdx.x;
  const int col = blockIdx.x;

  if (i < 27) {
    float phi = W[i * 3 + 0], th = W[i * 3 + 1], om = W[i * 3 + 2];
    float a = 0.5f * (phi + om), b = 0.5f * (phi - om);
    float c = cosf(0.5f * th), s = sinf(0.5f * th);
    float ca = cosf(a), sa = sinf(a), cb = cosf(b), sb = sinf(b);
    G[i][0] =  c * ca;  G[i][1] = -c * sa;   // g00 = ep*c
    G[i][2] = -s * cb;  G[i][3] = -s * sb;   // g01 = -conj(em)*s
    G[i][4] =  s * cb;  G[i][5] = -s * sb;   // g10 = em*s
    G[i][6] =  c * ca;  G[i][7] =  c * sa;   // g11 = conj(ep)*c
  }
  sr[i] = (i == col) ? 1.0f : 0.0f;
  si[i] = 0.0f;
  __syncthreads();

  for (int l = 0; l < 3; ++l) {
    for (int w = 0; w < 9; ++w) {
      const float* g = G[l * 9 + w];
      const int p = 8 - w, mask = 1 << p;
      const int b = (i >> p) & 1;
      const int i0 = i & ~mask, i1 = i | mask;
      float g0r = g[b * 4 + 0], g0i = g[b * 4 + 1];
      float g1r = g[b * 4 + 2], g1i = g[b * 4 + 3];
      float a0r = sr[i0], a0i = si[i0], a1r = sr[i1], a1i = si[i1];
      __syncthreads();
      sr[i] = g0r * a0r - g0i * a0i + g1r * a1r - g1i * a1i;
      si[i] = g0r * a0i + g0i * a0r + g1r * a1i + g1i * a1r;
      __syncthreads();
    }
    const int r = l % 8 + 1;
    for (int w = 0; w < 9; ++w) {
      const int pc = 8 - w, pt = 8 - ((w + r) % 9);
      const int src = ((i >> pc) & 1) ? (i ^ (1 << pt)) : i;
      float vr = sr[src], vi = si[src];
      __syncthreads();
      sr[i] = vr; si[i] = vi;
      __syncthreads();
    }
  }
  Urt_r[(size_t)col * DIM + i] = sr[i];
  Urt_i[(size_t)col * DIM + i] = si[i];
}

// ---------------------------------------------------------------------------
// Kernel 2: M = (D Ur)^T Ur + (D Ui)^T Ui   (512x512x512, f32 WMMA)
// 32x32 block tile, one 16x16 WMMA tile per wave -> 256 blocks / 1024 waves
// for latency hiding; k-loop unrolled 4x to keep ~16 L2 loads in flight.
// ---------------------------------------------------------------------------
#if __has_builtin(__builtin_amdgcn_wmma_f32_16x16x4_f32)
#define HAVE_WMMA_F32 1
#else
#warning "QNN marker: wmma f32 16x16x4 builtin MISSING - VALU fallback in use"
#endif

__global__ __launch_bounds__(128) void qnn_buildM(const float* __restrict__ Ur,
                                                  const float* __restrict__ Ui,
                                                  _Float16* __restrict__ Mh,
                                                  _Float16* __restrict__ Ml) {
  const int tid = threadIdx.x;
  const int lane = tid & 31, wave = tid >> 5;
  const int wm = wave & 1, wn = wave >> 1;
  const int row0 = blockIdx.x * 32 + wm * 16;
  const int col0 = blockIdx.y * 32 + wn * 16;
#if HAVE_WMMA_F32
  const int r = lane & 15;
  const int kb = (lane >> 4) * 2;
  const float* Ar = Ur + (size_t)(row0 + r) * DIM;
  const float* Ai = Ui + (size_t)(row0 + r) * DIM;
  const float* Br = Ur + (size_t)(col0 + r) * DIM;
  const float* Bi = Ui + (size_t)(col0 + r) * DIM;
  v8f C = {};
#pragma unroll 4
  for (int k = 0; k < DIM; k += 4) {
    const float s = (k < 256) ? 1.0f : -1.0f;   // d_i sign folded into A
    v2f ar = (*(const v2f*)(Ar + k + kb)) * s;
    v2f ai = (*(const v2f*)(Ai + k + kb)) * s;
    v2f br = *(const v2f*)(Br + k + kb);
    v2f bi = *(const v2f*)(Bi + k + kb);
    C = __builtin_amdgcn_wmma_f32_16x16x4_f32(false, ar, false, br, (short)0,
                                              C, false, false);
    C = __builtin_amdgcn_wmma_f32_16x16x4_f32(false, ai, false, bi, (short)0,
                                              C, false, false);
  }
  const int mhalf = (lane < 16) ? 0 : 8;
  const int colc = col0 + (lane & 15);
#pragma unroll
  for (int e = 0; e < 8; ++e) {
    const int rowc = row0 + e + mhalf;
    float f = C[e];
    _Float16 h = (_Float16)f;
    Mh[(size_t)rowc * DIM + colc] = h;
    Ml[(size_t)rowc * DIM + colc] = (_Float16)(f - (float)h);
  }
#else
  for (int q = tid; q < 32 * 32; q += 128) {
    const int rowc = blockIdx.x * 32 + q / 32;
    const int colc = blockIdx.y * 32 + q % 32;
    float acc = 0.f;
    for (int k = 0; k < DIM; ++k) {
      float d = (k < 256) ? 1.0f : -1.0f;
      acc += d * (Ur[(size_t)rowc * DIM + k] * Ur[(size_t)colc * DIM + k] +
                  Ui[(size_t)rowc * DIM + k] * Ui[(size_t)colc * DIM + k]);
    }
    _Float16 h = (_Float16)acc;
    Mh[(size_t)rowc * DIM + colc] = h;
    Ml[(size_t)rowc * DIM + colc] = (_Float16)(acc - (float)h);
  }
#endif
  (void)wn; (void)wm;
}

// ---------------------------------------------------------------------------
// Kernel 3: Y = X * M via split f16 WMMA (xh*Mh + xh*Ml + xl*Mh).
// Double-buffered LDS; B tiles via TDM (tensor_load_to_lds, TENSORcnt),
// A tiles via register-staged loads + f32->f16 hi/lo split.
// Fused deterministic epilogue -> xy_part & xx_part [sample][col_block]
// (no extra pass over X is needed anywhere else).
// ---------------------------------------------------------------------------
__global__ __launch_bounds__(128) void qnn_gemm(const float* __restrict__ X,
                                                const _Float16* __restrict__ Mh,
                                                const _Float16* __restrict__ Ml,
                                                float* __restrict__ xy_part,
                                                float* __restrict__ xx_part) {
  __shared__ __align__(16) _Float16 AhB[2][BM * LDA], AlB[2][BM * LDA];
  __shared__ __align__(16) _Float16 BhB[2][BN * LDB], BlB[2][BN * LDB];
  __shared__ float xyacc[2][BM], xxacc[2][BM];

  const int tid = threadIdx.x;
  const int lane = tid & 31, wave = tid >> 5;
  const int wm = wave & 1, wn = wave >> 1;
  const int bm0 = blockIdx.x * BM, bn0 = blockIdx.y * BN;

  const int m = tid >> 1, seg = tid & 1;   // staging coords (A and fallback-B)

  v8f C[2][2] = {};
  float4 areg[8];

  const int r = lane & 15;
  const int kbA = (lane < 16) ? 0 : 8;
  const int kbB = (lane < 16) ? 0 : 8;

  // ---- B tile staging: one TDM 2-D tile (16 rows x 64 halves) per wave ----
  auto issueB = [&](int k0, int buf) {
#if defined(HAVE_TDM)
    const int row0 = wave * 16;
    tdm_tile_load(lds_offset(&BhB[buf][row0 * LDB]),
                  Mh + (size_t)(bn0 + row0) * DIM + k0, 16);
    tdm_tile_load(lds_offset(&BlB[buf][row0 * LDB]),
                  Ml + (size_t)(bn0 + row0) * DIM + k0, 16);
#else
    const uint4* sh = (const uint4*)(Mh + (size_t)(bn0 + m) * DIM + k0 + seg * 32);
    const uint4* sl = (const uint4*)(Ml + (size_t)(bn0 + m) * DIM + k0 + seg * 32);
    uint4* dh = (uint4*)(&BhB[buf][m * LDB + seg * 32]);
    uint4* dl = (uint4*)(&BlB[buf][m * LDB + seg * 32]);
#pragma unroll
    for (int q = 0; q < 4; ++q) { dh[q] = sh[q]; dl[q] = sl[q]; }
#endif
  };

  auto loadA = [&](int k0) {
    const float* src = X + (size_t)(bm0 + m) * DIM + k0 + seg * 32;
#pragma unroll
    for (int q = 0; q < 8; ++q) areg[q] = ((const float4*)src)[q];
  };

  auto storeA = [&](int buf) {
    _Float16* dh = &AhB[buf][m * LDA + seg * 32];
    _Float16* dl = &AlB[buf][m * LDA + seg * 32];
#pragma unroll
    for (int q = 0; q < 8; ++q) {
      float4 f = areg[q];
      _Float16 h0 = (_Float16)f.x, h1 = (_Float16)f.y;
      _Float16 h2 = (_Float16)f.z, h3 = (_Float16)f.w;
      dh[q * 4 + 0] = h0; dh[q * 4 + 1] = h1; dh[q * 4 + 2] = h2; dh[q * 4 + 3] = h3;
      dl[q * 4 + 0] = (_Float16)(f.x - (float)h0);
      dl[q * 4 + 1] = (_Float16)(f.y - (float)h1);
      dl[q * 4 + 2] = (_Float16)(f.z - (float)h2);
      dl[q * 4 + 3] = (_Float16)(f.w - (float)h3);
    }
  };

  // ---- prologue: fill buffer 0 ----
  issueB(0, 0);
  loadA(0);
  storeA(0);
  WAIT_TENSORCNT0();
  __syncthreads();

  const int NSTAGE = DIM / BK;   // 8
  for (int s = 0; s < NSTAGE; ++s) {
    const int cur = s & 1, nxt = cur ^ 1;
    if (s + 1 < NSTAGE) {
      issueB((s + 1) * BK, nxt);   // async: overlaps with WMMA below
      loadA((s + 1) * BK);         // global loads in flight during compute
    }

    const _Float16* Ah = AhB[cur];
    const _Float16* Al = AlB[cur];
    const _Float16* Bh = BhB[cur];
    const _Float16* Bl = BlB[cur];
#pragma unroll
    for (int ks = 0; ks < BK; ks += 32) {
      F16Frag ah[2], al[2], bh[2], bl[2];
#pragma unroll
      for (int t = 0; t < 2; ++t) {
        const _Float16* pa = Ah + (wm * 32 + t * 16 + r) * LDA + ks + kbA;
        const _Float16* pl = Al + (wm * 32 + t * 16 + r) * LDA + ks + kbA;
        ah[t].h[0] = *(const v8h*)pa;  ah[t].h[1] = *(const v8h*)(pa + 16);
        al[t].h[0] = *(const v8h*)pl;  al[t].h[1] = *(const v8h*)(pl + 16);
        const _Float16* pb = Bh + (wn * 32 + t * 16 + r) * LDB + ks + kbB;
        const _Float16* pm = Bl + (wn * 32 + t * 16 + r) * LDB + ks + kbB;
        bh[t].h[0] = *(const v8h*)pb;  bh[t].h[1] = *(const v8h*)(pb + 16);
        bl[t].h[0] = *(const v8h*)pm;  bl[t].h[1] = *(const v8h*)(pm + 16);
      }
      // term-major order: consecutive WMMAs hit different accumulators
#pragma unroll
      for (int ti = 0; ti < 2; ++ti)
#pragma unroll
        for (int tj = 0; tj < 2; ++tj)
          C[ti][tj] = __builtin_amdgcn_wmma_f32_16x16x32_f16(
              false, ah[ti].v, false, bh[tj].v, (short)0, C[ti][tj], false, false);
#pragma unroll
      for (int ti = 0; ti < 2; ++ti)
#pragma unroll
        for (int tj = 0; tj < 2; ++tj)
          C[ti][tj] = __builtin_amdgcn_wmma_f32_16x16x32_f16(
              false, ah[ti].v, false, bl[tj].v, (short)0, C[ti][tj], false, false);
#pragma unroll
      for (int ti = 0; ti < 2; ++ti)
#pragma unroll
        for (int tj = 0; tj < 2; ++tj)
          C[ti][tj] = __builtin_amdgcn_wmma_f32_16x16x32_f16(
              false, al[ti].v, false, bh[tj].v, (short)0, C[ti][tj], false, false);
    }

    if (s + 1 < NSTAGE) {
      storeA(nxt);
      WAIT_TENSORCNT0();   // this wave's B quarter for stage s+1 has landed
    }
    __syncthreads();
  }

  // ---- fused epilogue ----
  // partial xy[m] = sum_n X[m][bn0+n] * Y[m][n]
  // partial xx[m] = sum_n X[m][bn0+n]^2       (each (m,n) touched exactly once)
  const int mhalf = (lane < 16) ? 0 : 8;
#pragma unroll
  for (int ti = 0; ti < 2; ++ti) {
#pragma unroll
    for (int e = 0; e < 8; ++e) {
      const int ml = wm * 32 + ti * 16 + e + mhalf;
      float accY = 0.f, accX = 0.f;
#pragma unroll
      for (int tj = 0; tj < 2; ++tj) {
        const int ng = bn0 + wn * 32 + tj * 16 + (lane & 15);
        float x = X[(size_t)(bm0 + ml) * DIM + ng];
        float vy = x * C[ti][tj][e];
        float vx = x * x;
        vy += __shfl_xor(vy, 1, 16);  vx += __shfl_xor(vx, 1, 16);
        vy += __shfl_xor(vy, 2, 16);  vx += __shfl_xor(vx, 2, 16);
        vy += __shfl_xor(vy, 4, 16);  vx += __shfl_xor(vx, 4, 16);
        vy += __shfl_xor(vy, 8, 16);  vx += __shfl_xor(vx, 8, 16);
        accY += vy;  accX += vx;      // every lane holds the 16-lane sums
      }
      if ((lane & 15) == 0) {         // unique (wn,row) writer
        xyacc[wn][ml] = accY;
        xxacc[wn][ml] = accX;
      }
    }
  }
  __syncthreads();
  if (tid < BM) {
    xy_part[(size_t)(bm0 + tid) * 8 + blockIdx.y] = xyacc[0][tid] + xyacc[1][tid];
    xx_part[(size_t)(bm0 + tid) * 8 + blockIdx.y] = xxacc[0][tid] + xxacc[1][tid];
  }
}

// ---------------------------------------------------------------------------
// Kernel 4: out[s] = (sum_c xy_part[s][c]) / (sum_c xx_part[s][c])
// One thread per sample; 8 contiguous floats per sample in each array.
// ---------------------------------------------------------------------------
__global__ __launch_bounds__(256) void qnn_out(const float* __restrict__ xy_part,
                                               const float* __restrict__ xx_part,
                                               float* __restrict__ out) {
  const int s = blockIdx.x * 256 + threadIdx.x;
  const float4* py = (const float4*)(xy_part + (size_t)s * 8);
  const float4* px = (const float4*)(xx_part + (size_t)s * 8);
  float4 y0 = py[0], y1 = py[1];
  float4 x0 = px[0], x1 = px[1];
  float xy = (y0.x + y0.y) + (y0.z + y0.w) + (y1.x + y1.y) + (y1.z + y1.w);
  float xx = (x0.x + x0.y) + (x0.z + x0.w) + (x1.x + x1.y) + (x1.z + x1.w);
  out[s] = xy / xx;
}

// ---------------------------------------------------------------------------
extern "C" void kernel_launch(void* const* d_in, const int* in_sizes, int n_in,
                              void* d_out, int out_size, void* d_ws, size_t ws_size,
                              hipStream_t stream) {
  const float* X = (const float*)d_in[0];     // 32768 x 512 f32
  const float* W = (const float*)d_in[1];     // 3 x 9 x 3 f32
  float* out = (float*)d_out;                 // 32768 f32

  char* ws = (char*)d_ws;
  float*     Ur      = (float*)(ws);                                // 1 MB
  float*     Ui      = (float*)(ws + (1u << 20));                   // 1 MB
  _Float16*  Mh      = (_Float16*)(ws + (2u << 20));                // 512 KB
  _Float16*  Ml      = (_Float16*)(ws + (2u << 20) + (512u << 10)); // 512 KB
  float*     xy_part = (float*)(ws + (3u << 20));                   // 1 MB
  float*     xx_part = (float*)(ws + (4u << 20));                   // 1 MB

  qnn_buildU<<<DIM, DIM, 0, stream>>>(W, Ur, Ui);
  qnn_buildM<<<dim3(16, 16), 128, 0, stream>>>(Ur, Ui, Mh, Ml);
  qnn_gemm<<<dim3(32768 / BM, DIM / BN), 128, 0, stream>>>(X, Mh, Ml,
                                                           xy_part, xx_part);
  qnn_out<<<32768 / 256, 256, 0, stream>>>(xy_part, xx_part, out);
}
